// SPOT_CA_0_31172872634543
// MI455X (gfx1250) — compile-verified, matrix operand order checked by reading, not run on hardware
//
#include <hip/hip_runtime.h>
#include <hip/hip_bf16.h>
#include <stdint.h>

#define QL    2048
#define KLEN  4096
#define CDIM  256
#define NHEAD 8
#define HDIM  32
#define KTOP  410            // max(32, ceil(0.1*4096))
#define LN_EPS 1e-5f
#define NSUB  4              // N-tiles per wave in the WMMA GEMM (16x64 output)

typedef __bf16 bf16x16 __attribute__((ext_vector_type(16)));
typedef float  f32x8   __attribute__((ext_vector_type(8)));
typedef unsigned int v4u __attribute__((ext_vector_type(4)));
typedef int v8i __attribute__((ext_vector_type(8)));
typedef int v4i __attribute__((ext_vector_type(4)));

__device__ __forceinline__ unsigned short f2bf_rne(float f) {
    unsigned int u = __builtin_bit_cast(unsigned int, f);
    unsigned int r = u + 0x7FFFu + ((u >> 16) & 1u);
    return (unsigned short)(r >> 16);
}
// monotonic float -> uint ordering
__device__ __forceinline__ unsigned int ord_of(float f) {
    unsigned int u = __builtin_bit_cast(unsigned int, f);
    return (u & 0x80000000u) ? ~u : (u | 0x80000000u);
}

// ---- Tensor Data Mover: DMA one contiguous row of f32 from global into LDS ----
// D# packing per CDNA5 ISA 8.3/8.4: group0 {count=1, lds_addr, global_addr, type=2},
// group1 {data_size=4B, tensor_dim0=n, tensor_dim1=1, tile_dim0=n, stride0=n}.
// This toolchain exposes the 6-arg builtin: (g0, g1, g2, g3, g_extra, cpol).
#if __has_builtin(__builtin_amdgcn_tensor_load_to_lds)
#define HAVE_TDM 1
__device__ __forceinline__ void tdm_load_row_f32(unsigned lds_off, const float* gptr, unsigned n) {
    unsigned long long ga = (unsigned long long)(uintptr_t)gptr;
    v4u g0;
    g0.x = 1u;                                             // count=1, user descriptor
    g0.y = lds_off;                                        // lds_addr (bytes)
    g0.z = (unsigned)(ga & 0xFFFFFFFFu);                   // global_addr[31:0]
    g0.w = (unsigned)((ga >> 32) & 0x01FFFFFFu) | (2u << 30); // addr[56:32] | type=2
    v8i g1;
    g1[0] = (int)(2u << 16);                               // data_size code 2 (4 bytes)
    g1[1] = (int)((n & 0xFFFFu) << 16);                    // tensor_dim0[15:0]
    g1[2] = (int)(((n >> 16) & 0xFFFFu) | (1u << 16));     // tensor_dim0[31:16] | tensor_dim1=1
    g1[3] = (int)((n & 0xFFFFu) << 16);                    // tile_dim0 = n
    g1[4] = 0;                                             // tile_dim1=0, tile_dim2=0 (unused)
    g1[5] = (int)n;                                        // tensor_dim0_stride[31:0]
    g1[6] = 0;
    g1[7] = 0;
    v4i z4 = {0, 0, 0, 0};
    v8i z8 = {0, 0, 0, 0, 0, 0, 0, 0};
    __builtin_amdgcn_tensor_load_to_lds(g0, g1, z4, z4, z8, 0);
    __builtin_amdgcn_s_wait_tensorcnt(0);
}
#else
#define HAVE_TDM 0
#endif

// ---------------- add positional + LayerNorm -> bf16 ----------------
__global__ void __launch_bounds__(256)
add_ln_bf16_kernel(const float* __restrict__ x, const float* __restrict__ pos,
                   const float* __restrict__ g, const float* __restrict__ b,
                   unsigned short* __restrict__ out_bf) {
    int row = blockIdx.x, c = threadIdx.x;               // 256 threads == CDIM
    __shared__ float red[256];
    float v = x[(size_t)row * CDIM + c] + pos[(size_t)row * CDIM + c];
    red[c] = v; __syncthreads();
    for (int s = 128; s > 0; s >>= 1) { if (c < s) red[c] += red[c + s]; __syncthreads(); }
    float mu = red[0] * (1.0f / CDIM);
    __syncthreads();
    float dv = v - mu;
    red[c] = dv * dv; __syncthreads();
    for (int s = 128; s > 0; s >>= 1) { if (c < s) red[c] += red[c + s]; __syncthreads(); }
    float var = red[0] * (1.0f / CDIM);
    float y = dv * rsqrtf(var + LN_EPS) * g[c] + b[c];
    out_bf[(size_t)row * CDIM + c] = f2bf_rne(y);
}

// ---------------- f32 -> bf16 convert (weights) ----------------
__global__ void f32_to_bf16_kernel(const float* __restrict__ a,
                                   unsigned short* __restrict__ o, int n) {
    int i = blockIdx.x * 256 + threadIdx.x;
    if (i < n) o[i] = f2bf_rne(a[i]);
}

// ---------------- elementwise product -> bf16 ----------------
__global__ void mul_to_bf16_kernel(const float* __restrict__ a, const float* __restrict__ b,
                                   unsigned short* __restrict__ o, int n) {
    int i = blockIdx.x * 256 + threadIdx.x;
    if (i < n) o[i] = f2bf_rne(a[i] * b[i]);
}

// ---------------- per-head L2 normalize -> bf16 [H][M][32] ----------------
__global__ void __launch_bounds__(256)
l2norm_heads_kernel(const float* __restrict__ X, unsigned short* __restrict__ out, int M) {
    int row = blockIdx.x, tid = threadIdx.x;
    int h = tid >> 5, d = tid & 31;                      // one wave32 per head
    float v = X[(size_t)row * CDIM + h * HDIM + d];
    float ss = v * v;
    #pragma unroll
    for (int m = 16; m; m >>= 1) ss += __shfl_xor(ss, m, 32);
    float n = fmaxf(sqrtf(ss), 1e-12f);
    out[((size_t)h * M + row) * HDIM + d] = f2bf_rne(v / n);
}

// ------- WMMA GEMM: Y[M,N] = A[M,K] * Bw[N,K]^T (+bias)(+resid) -------
// bf16 inputs, f32 accumulate. One wave computes a 16x64 output strip:
// A fragment loaded once per k-step, reused across NSUB=4 WMMAs
// (4 independent accumulator chains for the scheduler to overlap with loads).
__global__ void __launch_bounds__(32)
wmma_gemm_bf16(const unsigned short* __restrict__ A, const unsigned short* __restrict__ Bw,
               const float* __restrict__ bias, const float* __restrict__ resid,
               float* __restrict__ Y, int M, int N, int K) {
    int tileM  = blockIdx.x;
    int tileN0 = blockIdx.y * NSUB;
    int lane = threadIdx.x & 31;
    int half = lane >> 4;                                // K-half select (ISA A layout)
    int r    = lane & 15;                                // A row / B column
    const unsigned short* arow = A + (size_t)(tileM * 16 + r) * K;
    const unsigned short* brow[NSUB];
    #pragma unroll
    for (int t = 0; t < NSUB; ++t)
        brow[t] = Bw + (size_t)((tileN0 + t) * 16 + r) * K;
    f32x8 acc[NSUB] = {};
    union frag_u { bf16x16 v; unsigned int u[8]; };
    for (int kb = 0; kb < K; kb += 32) {
        int k0 = kb + half * 8;
        frag_u af;
        #pragma unroll
        for (int j = 0; j < 4; ++j) {
            af.u[j]     = *(const unsigned int*)(arow + k0 + 2 * j);
            af.u[j + 4] = *(const unsigned int*)(arow + k0 + 16 + 2 * j);
        }
        if (kb + 32 < K) __builtin_prefetch(arow + kb + 32, 0, 0);  // global_prefetch_b8
        #pragma unroll
        for (int t = 0; t < NSUB; ++t) {
            frag_u bfr;
            #pragma unroll
            for (int j = 0; j < 4; ++j) {
                bfr.u[j]     = *(const unsigned int*)(brow[t] + k0 + 2 * j);
                bfr.u[j + 4] = *(const unsigned int*)(brow[t] + k0 + 16 + 2 * j);
            }
            if (kb + 32 < K) __builtin_prefetch(brow[t] + kb + 32, 0, 0);
            acc[t] = __builtin_amdgcn_wmma_f32_16x16x32_bf16(
                         false, af.v, false, bfr.v, (short)0, acc[t], false, false);
        }
    }
    #pragma unroll
    for (int t = 0; t < NSUB; ++t) {
        int col = (tileN0 + t) * 16 + r;
        float bv = bias ? bias[col] : 0.0f;
        #pragma unroll
        for (int i = 0; i < 8; ++i) {
            int rowi = tileM * 16 + half * 8 + i;        // C/D layout: lanes 16-31 -> M+8
            size_t idx = (size_t)rowi * N + col;
            float o = acc[t][i] + bv;
            if (resid) o += resid[idx];
            Y[idx] = o;
        }
    }
}

// ------ per (head,q) row: top-KTOP threshold select + softmax + weighted V gather ------
__global__ void __launch_bounds__(256)
topk_softmax_gather_kernel(const float* __restrict__ sim,
                           const float* __restrict__ key, const float* __restrict__ key_pos,
                           int h, float* __restrict__ merged) {
    int q = blockIdx.x, tid = threadIdx.x;
    __shared__ float row[KLEN];                          // 16 KB, LDS-resident row
    __shared__ float red[256];
    __shared__ int   cnt;
    __shared__ float part[8][HDIM];
    __shared__ float sden;
    const float* srow = sim + (size_t)q * KLEN;
#if HAVE_TDM
    // wave 0 DMAs the whole sim row into LDS via the Tensor Data Mover
    if (tid < 32) {
        tdm_load_row_f32((unsigned)(uintptr_t)&row[0], srow, KLEN);
    }
#else
    for (int i = tid; i < KLEN; i += 256) row[i] = srow[i];
#endif
    __syncthreads();
    // row max / min
    float lmax = -__builtin_inff(), lmin = __builtin_inff();
    for (int i = tid; i < KLEN; i += 256) { float x = row[i]; lmax = fmaxf(lmax, x); lmin = fminf(lmin, x); }
    red[tid] = lmax; __syncthreads();
    for (int s = 128; s; s >>= 1) { if (tid < s) red[tid] = fmaxf(red[tid], red[tid + s]); __syncthreads(); }
    float m = red[0]; __syncthreads();
    red[tid] = lmin; __syncthreads();
    for (int s = 128; s; s >>= 1) { if (tid < s) red[tid] = fminf(red[tid], red[tid + s]); __syncthreads(); }
    float mn = red[0]; __syncthreads();
    // binary search for the KTOP-th largest value (ordered-uint space)
    unsigned long long lo = ord_of(mn);                  // count(>=lo) == KLEN >= KTOP
    unsigned long long hi = (unsigned long long)ord_of(m) + 1ull;  // count(>=hi) == 0
    while (hi - lo > 1ull) {
        unsigned int mid = (unsigned int)((lo + hi) >> 1);
        if (tid == 0) cnt = 0;
        __syncthreads();
        int c = 0;
        for (int i = tid; i < KLEN; i += 256) if (ord_of(row[i]) >= mid) ++c;
        atomicAdd(&cnt, c);                              // ds_add_u32
        __syncthreads();
        if (cnt >= KTOP) lo = mid; else hi = mid;        // uniform across block
        __syncthreads();
    }
    unsigned int thr = (unsigned int)lo;
    // softmax weights for kept set (written back into the LDS row)
    if (tid == 0) sden = 0.0f;
    __syncthreads();
    const float scale = 0.17677669529663687f;            // 32^-0.5
    float dl = 0.0f;
    for (int i = tid; i < KLEN; i += 256) {
        float x = row[i];
        float w = (ord_of(x) >= thr) ? __expf((x - m) * scale) : 0.0f;
        row[i] = w; dl += w;
    }
    atomicAdd(&sden, dl);                                // ds_add_f32
    __syncthreads();
    float den = sden;
    // weighted sum of V rows (V = key + key_pos, computed on the fly)
    int d = tid & 31, slice = tid >> 5;
    int base = h * HDIM + d;
    float acc = 0.0f;
    for (int k = slice * (KLEN / 8); k < (slice + 1) * (KLEN / 8); ++k) {
        float w = row[k];
        if (w != 0.0f)
            acc += w * (key[(size_t)k * CDIM + base] + key_pos[(size_t)k * CDIM + base]);
    }
    part[slice][d] = acc;
    __syncthreads();
    if (slice == 0) {
        float s = 0.0f;
        #pragma unroll
        for (int j = 0; j < 8; ++j) s += part[j][d];
        merged[(size_t)q * CDIM + base] = s / den;
    }
}

// ---------------- column (over Q) L2 norm, *alpha, +merged -> bf16 ----------------
__global__ void __launch_bounds__(256)
colnorm_alpha_kernel(const float* __restrict__ inter, const float* __restrict__ merged,
                     const float* __restrict__ alpha, unsigned short* __restrict__ out_bf) {
    int c = blockIdx.x, tid = threadIdx.x;
    __shared__ float red[256];
    float ss = 0.0f;
    for (int q = tid; q < QL; q += 256) { float x = inter[(size_t)q * CDIM + c]; ss += x * x; }
    red[tid] = ss; __syncthreads();
    for (int s = 128; s; s >>= 1) { if (tid < s) red[tid] += red[tid + s]; __syncthreads(); }
    float inv = 1.0f / fmaxf(sqrtf(red[0]), 1e-12f);
    float al = alpha[c];
    for (int q = tid; q < QL; q += 256) {
        size_t idx = (size_t)q * CDIM + c;
        out_bf[idx] = f2bf_rne(inter[idx] * inv * al + merged[idx]);
    }
}

extern "C" void kernel_launch(void* const* d_in, const int* in_sizes, int n_in,
                              void* d_out, int out_size, void* d_ws, size_t ws_size,
                              hipStream_t stream) {
    (void)in_sizes; (void)n_in; (void)out_size; (void)ws_size;
    const float* query   = (const float*)d_in[0];
    const float* key     = (const float*)d_in[1];
    const float* qpos    = (const float*)d_in[2];
    const float* kpos    = (const float*)d_in[3];
    const float* q_ln_g  = (const float*)d_in[4];
    const float* q_ln_b  = (const float*)d_in[5];
    const float* q_w     = (const float*)d_in[6];
    const float* q_b     = (const float*)d_in[7];
    const float* k_ln_g  = (const float*)d_in[8];
    const float* k_ln_b  = (const float*)d_in[9];
    const float* k_w     = (const float*)d_in[10];
    const float* k_b     = (const float*)d_in[11];
    const float* proj_w  = (const float*)d_in[12];
    const float* proj_b  = (const float*)d_in[13];
    const float* final_w = (const float*)d_in[14];
    const float* final_b = (const float*)d_in[15];
    const float* alpha   = (const float*)d_in[16];
    float* out = (float*)d_out;

    char* ws = (char*)d_ws;
    size_t off = 0;
    auto alloc = [&](size_t bytes) -> char* {
        char* p = ws + off; off = (off + bytes + 255) & ~(size_t)255; return p;
    };
    unsigned short* qln    = (unsigned short*)alloc((size_t)QL * CDIM * 2);
    unsigned short* kln    = (unsigned short*)alloc((size_t)KLEN * CDIM * 2);
    unsigned short* wq     = (unsigned short*)alloc((size_t)CDIM * CDIM * 2);
    unsigned short* wk     = (unsigned short*)alloc((size_t)CDIM * CDIM * 2);
    unsigned short* wproj  = (unsigned short*)alloc((size_t)CDIM * CDIM * 2);
    unsigned short* wfinal = (unsigned short*)alloc((size_t)CDIM * CDIM * 2);
    float*          qproj  = (float*)alloc((size_t)QL * CDIM * 4);
    float*          kproj  = (float*)alloc((size_t)KLEN * CDIM * 4);
    unsigned short* qhat   = (unsigned short*)alloc((size_t)QL * CDIM * 2);   // [H][QL][32]
    unsigned short* khat   = (unsigned short*)alloc((size_t)KLEN * CDIM * 2); // [H][KL][32]
    float*          sim    = (float*)alloc((size_t)QL * KLEN * 4);            // per-head reuse
    float*          merged = (float*)alloc((size_t)QL * CDIM * 4);
    unsigned short* prod   = (unsigned short*)alloc((size_t)QL * CDIM * 2);
    float*          inter  = (float*)alloc((size_t)QL * CDIM * 4);
    unsigned short* outbf  = (unsigned short*)alloc((size_t)QL * CDIM * 2);

    // 1) add pos + LN -> bf16
    add_ln_bf16_kernel<<<QL, 256, 0, stream>>>(query, qpos, q_ln_g, q_ln_b, qln);
    add_ln_bf16_kernel<<<KLEN, 256, 0, stream>>>(key, kpos, k_ln_g, k_ln_b, kln);

    // 2) weights -> bf16
    const int WN = CDIM * CDIM;
    f32_to_bf16_kernel<<<(WN + 255) / 256, 256, 0, stream>>>(q_w, wq, WN);
    f32_to_bf16_kernel<<<(WN + 255) / 256, 256, 0, stream>>>(k_w, wk, WN);
    f32_to_bf16_kernel<<<(WN + 255) / 256, 256, 0, stream>>>(proj_w, wproj, WN);
    f32_to_bf16_kernel<<<(WN + 255) / 256, 256, 0, stream>>>(final_w, wfinal, WN);

    // 3) q/k projections (WMMA, 16x64 strip per wave)
    wmma_gemm_bf16<<<dim3(QL / 16, CDIM / (16 * NSUB)), 32, 0, stream>>>(
        qln, wq, q_b, nullptr, qproj, QL, CDIM, CDIM);
    wmma_gemm_bf16<<<dim3(KLEN / 16, CDIM / (16 * NSUB)), 32, 0, stream>>>(
        kln, wk, k_b, nullptr, kproj, KLEN, CDIM, CDIM);

    // 4) per-head L2 normalize -> bf16 [H][M][32]
    l2norm_heads_kernel<<<QL, 256, 0, stream>>>(qproj, qhat, QL);
    l2norm_heads_kernel<<<KLEN, 256, 0, stream>>>(kproj, khat, KLEN);

    // 5) per head: sim GEMM (WMMA, K=32) then threshold-select + softmax + gather
    for (int h = 0; h < NHEAD; ++h) {
        wmma_gemm_bf16<<<dim3(QL / 16, KLEN / (16 * NSUB)), 32, 0, stream>>>(
            qhat + (size_t)h * QL * HDIM, khat + (size_t)h * KLEN * HDIM,
            nullptr, nullptr, sim, QL, KLEN, HDIM);
        topk_softmax_gather_kernel<<<QL, 256, 0, stream>>>(sim, key, kpos, h, merged);
    }

    // 6) interaction = (merged * qproj) @ proj_w.T + proj_b
    mul_to_bf16_kernel<<<(QL * CDIM + 255) / 256, 256, 0, stream>>>(merged, qproj, prod, QL * CDIM);
    wmma_gemm_bf16<<<dim3(QL / 16, CDIM / (16 * NSUB)), 32, 0, stream>>>(
        prod, wproj, proj_b, nullptr, inter, QL, CDIM, CDIM);

    // 7) column-wise (over Q) L2 norm * alpha + merged -> bf16
    colnorm_alpha_kernel<<<CDIM, 256, 0, stream>>>(inter, merged, alpha, outbf);

    // 8) final GEMM + bias + residual -> d_out
    wmma_gemm_bf16<<<dim3(QL / 16, CDIM / (16 * NSUB)), 32, 0, stream>>>(
        outbf, wfinal, final_b, query, out, QL, CDIM, CDIM);
}
